// LSTMCell_28475633172494
// MI455X (gfx1250) — matrix operand order repaired and checked by reading, not compiled, not run on hardware
//
#include <hip/hip_runtime.h>
#include <hip/hip_bf16.h>
#include <cstdint>

// LSTM forward, B=64 T=512 I=1024 H=1024.
// Phase 1: xW[t][b][n] = x[b,t,:] . Wi[n,:]  (bf16-split WMMA GEMM)
// Phase 2: per t: preact += h . Wh^T (split-K WMMA + f32 atomics), then gates.

static constexpr int kB   = 64;
static constexpr int kT   = 512;
static constexpr int kI   = 1024;
static constexpr int kH   = 1024;
static constexpr int k4H  = 4096;
static constexpr int kM   = kB * kT;   // 32768

typedef __attribute__((ext_vector_type(16))) __bf16 v16bf;
typedef __attribute__((ext_vector_type(8)))  __bf16 v8bf;
typedef __attribute__((ext_vector_type(8)))  float  v8f;

static __device__ __forceinline__ v8f wmma_bf16(v16bf a, v16bf b, v8f c) {
  // D = A*B + C, fp32 accumulate, 16x16x32 bf16
  return __builtin_amdgcn_wmma_f32_16x16x32_bf16(
      /*neg_a=*/false, a, /*neg_b=*/false, b,
      /*c_mod=*/(short)0, c, /*reuse_a=*/false, /*reuse_b=*/false);
}

// Assemble a 16-half fragment from two 16-byte LDS chunks.
static __device__ __forceinline__ v16bf ldfrag(const __bf16* p0, const __bf16* p1) {
  v16bf r;
  ((v8bf*)&r)[0] = *(const v8bf*)p0;
  ((v8bf*)&r)[1] = *(const v8bf*)p1;
  return r;
}

static __device__ __forceinline__ float sigf(float x) {
  return 1.0f / (1.0f + __expf(-x));
}

// ---- CDNA5 async global->LDS path (ASYNCcnt), guarded for toolchain drift.
#if defined(__has_builtin)
#if __has_builtin(__builtin_amdgcn_global_load_async_to_lds_b128) && \
    __has_builtin(__builtin_amdgcn_s_wait_asynccnt)
#define USE_ASYNC_LDS 1
#endif
#endif
#ifndef USE_ASYNC_LDS
#define USE_ASYNC_LDS 0
#endif

#if USE_ASYNC_LDS
// Builtin signature (from hipcc diagnostic): param0 = vector_size(16) int
// pointer in the global address space, param1 = same in LDS address space,
// then two immediate ints (offset, cpol).
typedef int v4i_gcc __attribute__((vector_size(16)));
typedef __attribute__((address_space(1))) v4i_gcc* gvec_p;
typedef __attribute__((address_space(3))) v4i_gcc* lvec_p;

static __device__ __forceinline__ void async_cp16(void* lds_dst, const void* gsrc) {
  // Per-lane: LDS[lds_dst] = MEM[gsrc], 16 bytes, tracked by ASYNCcnt.
  __builtin_amdgcn_global_load_async_to_lds_b128(
      (gvec_p)(uintptr_t)gsrc,
      (lvec_p)(uint32_t)(uintptr_t)lds_dst,   // generic LDS ptr low 32b = LDS offset
      /*offset=*/0, /*cpol=*/0);
}
#endif

// ---------------------------------------------------------------- converts
__global__ void k_split(const float* __restrict__ in,
                        __bf16* __restrict__ hi, __bf16* __restrict__ lo, int n) {
  int i = blockIdx.x * blockDim.x + threadIdx.x;
  if (i < n) {
    float v = in[i];
    __bf16 h = (__bf16)v;
    hi[i] = h;
    lo[i] = (__bf16)(v - (float)h);
  }
}

__global__ void k_init_state(__bf16* __restrict__ hh, __bf16* __restrict__ hl,
                             float* __restrict__ c, int n) {
  int i = blockIdx.x * blockDim.x + threadIdx.x;
  if (i < n) { hh[i] = (__bf16)0.0f; hl[i] = (__bf16)0.0f; c[i] = 0.0f; }
}

// ---------------------------------------------------------------- phase 1
// C[m][n] = sum_k x[m][k]*Wi[n][k], bf16 split (hi*hi + hi*lo + lo*hi).
// Block tile 128x128, 4 waves in 2x2 grid, wave tile 64x64, K-step 32.
// Async double-buffered LDS staging; stores scattered so xW is [t][b][n].
__global__ __launch_bounds__(128)
void k_gemm_xWi(const __bf16* __restrict__ xh, const __bf16* __restrict__ xl,
                const __bf16* __restrict__ wih, const __bf16* __restrict__ wil,
                float* __restrict__ xW) {
  constexpr int LS = 40;  // padded halves per 32-half row (80B, 16B-aligned)
  constexpr int NK = kI / 32;

  const int tid  = threadIdx.x;
  const int lane = tid & 31;
  const int wave = tid >> 5;
  const int wm   = (wave >> 1) * 64;   // wave row offset in block tile
  const int wn   = (wave & 1) * 64;    // wave col offset
  const int m0   = blockIdx.x * 128;
  const int n0   = blockIdx.y * 128;

  const int lh   = lane & 15;
  const int kbA  = (lane < 16) ? 0 : 16;   // A: low lanes K0..15, high K16..31
  const int kbB  = (lane < 16) ? 0 : 16;   // B: low lanes K0..15, high K16..31
  const int rA   = (lane < 16) ? 0 : 8;    // C/D: high lanes hold M+8

  v8f acc[4][4] = {};

  auto compute = [&](const __bf16* pAh, const __bf16* pAl,
                     const __bf16* pBh, const __bf16* pBl) {
    v16bf ah[4], al[4], bh[4], bl[4];
    #pragma unroll
    for (int i = 0; i < 4; ++i) {
      int row = (wm + i * 16 + lh) * LS;
      ah[i] = ldfrag(&pAh[row + kbA], &pAh[row + kbA + 8]);
      al[i] = ldfrag(&pAl[row + kbA], &pAl[row + kbA + 8]);
    }
    #pragma unroll
    for (int j = 0; j < 4; ++j) {
      int row = (wn + j * 16 + lh) * LS;
      bh[j] = ldfrag(&pBh[row + kbB], &pBh[row + kbB + 8]);
      bl[j] = ldfrag(&pBl[row + kbB], &pBl[row + kbB + 8]);
    }
    #pragma unroll
    for (int i = 0; i < 4; ++i) {
      #pragma unroll
      for (int j = 0; j < 4; ++j) {
        acc[i][j] = wmma_bf16(ah[i], bh[j], acc[i][j]);
        acc[i][j] = wmma_bf16(ah[i], bl[j], acc[i][j]);
        acc[i][j] = wmma_bf16(al[i], bh[j], acc[i][j]);
      }
    }
  };

#if USE_ASYNC_LDS
  __shared__ __align__(16) __bf16 sAh[2][128 * LS], sAl[2][128 * LS];
  __shared__ __align__(16) __bf16 sBh[2][128 * LS], sBl[2][128 * LS];

  auto stage = [&](int buf, int k0) {     // 16 async b128 ops per thread
    const __bf16* gAh = xh  + (size_t)(m0 + tid) * kI + k0;
    const __bf16* gAl = xl  + (size_t)(m0 + tid) * kI + k0;
    const __bf16* gBh = wih + (size_t)(n0 + tid) * kI + k0;
    const __bf16* gBl = wil + (size_t)(n0 + tid) * kI + k0;
    #pragma unroll
    for (int c = 0; c < 4; ++c) {
      async_cp16(&sAh[buf][tid * LS + c * 8], gAh + c * 8);
      async_cp16(&sAl[buf][tid * LS + c * 8], gAl + c * 8);
      async_cp16(&sBh[buf][tid * LS + c * 8], gBh + c * 8);
      async_cp16(&sBl[buf][tid * LS + c * 8], gBl + c * 8);
    }
  };

  stage(0, 0);
  for (int kk = 0; kk < NK; ++kk) {
    const int buf = kk & 1;
    __syncthreads();                         // prior compute done in all waves
    if (kk + 1 < NK) {
      stage(buf ^ 1, (kk + 1) * 32);         // DMA next tile during compute
      __builtin_amdgcn_s_wait_asynccnt(16);  // my tile-kk ops have landed
    } else {
      __builtin_amdgcn_s_wait_asynccnt(0);
    }
    __syncthreads();                         // everyone's tile-kk data visible
    compute(sAh[buf], sAl[buf], sBh[buf], sBl[buf]);
  }
#else
  __shared__ __align__(16) __bf16 sAh[128 * LS], sAl[128 * LS];
  __shared__ __align__(16) __bf16 sBh[128 * LS], sBl[128 * LS];

  for (int k0 = 0; k0 < kI; k0 += 32) {
    __syncthreads();
    {
      const __bf16* gAh = xh  + (size_t)(m0 + tid) * kI + k0;
      const __bf16* gAl = xl  + (size_t)(m0 + tid) * kI + k0;
      const __bf16* gBh = wih + (size_t)(n0 + tid) * kI + k0;
      const __bf16* gBl = wil + (size_t)(n0 + tid) * kI + k0;
      #pragma unroll
      for (int c = 0; c < 4; ++c) {
        *(uint4*)&sAh[tid * LS + c * 8] = *(const uint4*)(gAh + c * 8);
        *(uint4*)&sAl[tid * LS + c * 8] = *(const uint4*)(gAl + c * 8);
        *(uint4*)&sBh[tid * LS + c * 8] = *(const uint4*)(gBh + c * 8);
        *(uint4*)&sBl[tid * LS + c * 8] = *(const uint4*)(gBl + c * 8);
      }
    }
    __syncthreads();
    compute(sAh, sAl, sBh, sBl);
  }
#endif

  #pragma unroll
  for (int i = 0; i < 4; ++i) {
    #pragma unroll
    for (int j = 0; j < 4; ++j) {
      #pragma unroll
      for (int r = 0; r < 8; ++r) {
        int m = m0 + wm + i * 16 + rA + r;       // m = b*T + t
        int n = n0 + wn + j * 16 + lh;
        int b = m >> 9;                           // /512
        int t = m & 511;
        xW[((size_t)t * kB + b) * k4H + n] = acc[i][j][r];
      }
    }
  }
}

// ---------------------------------------------------------------- phase 2a
// preact[b][n] += sum_{k in chunk} h[b][k]*Wh[n][k], split-K over blockIdx.y.
// Block: M=64 rows, N=128 cols, K-chunk 128. 4 waves, each 16x128.
__global__ __launch_bounds__(128)
void k_step_hWh(const __bf16* __restrict__ hh, const __bf16* __restrict__ hl,
                const __bf16* __restrict__ whh, const __bf16* __restrict__ whl,
                float* __restrict__ pre) {
  constexpr int LS = 40;
  constexpr int NK = 4;

  const int tid  = threadIdx.x;
  const int lane = tid & 31;
  const int wave = tid >> 5;
  const int mb   = wave * 16;               // wave handles rows mb..mb+15
  const int n0   = blockIdx.x * 128;
  const int kch  = blockIdx.y * 128;

  const int lh  = lane & 15;
  const int kbA = (lane < 16) ? 0 : 16;
  const int kbB = (lane < 16) ? 0 : 16;
  const int rA  = (lane < 16) ? 0 : 8;

  v8f acc[8] = {};

  auto compute = [&](const __bf16* pAh, const __bf16* pAl,
                     const __bf16* pBh, const __bf16* pBl) {
    const int arow = (mb + lh) * LS;
    v16bf ah = ldfrag(&pAh[arow + kbA], &pAh[arow + kbA + 8]);
    v16bf al = ldfrag(&pAl[arow + kbA], &pAl[arow + kbA + 8]);
    #pragma unroll
    for (int j = 0; j < 8; ++j) {
      int brow = (j * 16 + lh) * LS;
      v16bf bh = ldfrag(&pBh[brow + kbB], &pBh[brow + kbB + 8]);
      v16bf bl = ldfrag(&pBl[brow + kbB], &pBl[brow + kbB + 8]);
      acc[j] = wmma_bf16(ah, bh, acc[j]);
      acc[j] = wmma_bf16(ah, bl, acc[j]);
      acc[j] = wmma_bf16(al, bh, acc[j]);
    }
  };

#if USE_ASYNC_LDS
  __shared__ __align__(16) __bf16 sAh[2][64 * LS], sAl[2][64 * LS];
  __shared__ __align__(16) __bf16 sBh[2][128 * LS], sBl[2][128 * LS];

  auto stage = [&](int buf, int k0) {     // 12 async b128 ops per thread
    for (int it = tid; it < 256; it += 128) {
      int r = it >> 2, c = (it & 3) * 8;
      async_cp16(&sAh[buf][r * LS + c], &hh[(size_t)r * kH + k0 + c]);
      async_cp16(&sAl[buf][r * LS + c], &hl[(size_t)r * kH + k0 + c]);
    }
    for (int it = tid; it < 512; it += 128) {
      int r = it >> 2, c = (it & 3) * 8;
      async_cp16(&sBh[buf][r * LS + c], &whh[(size_t)(n0 + r) * kH + k0 + c]);
      async_cp16(&sBl[buf][r * LS + c], &whl[(size_t)(n0 + r) * kH + k0 + c]);
    }
  };

  stage(0, kch);
  for (int kk = 0; kk < NK; ++kk) {
    const int buf = kk & 1;
    __syncthreads();
    if (kk + 1 < NK) {
      stage(buf ^ 1, kch + (kk + 1) * 32);
      __builtin_amdgcn_s_wait_asynccnt(12);
    } else {
      __builtin_amdgcn_s_wait_asynccnt(0);
    }
    __syncthreads();
    compute(sAh[buf], sAl[buf], sBh[buf], sBl[buf]);
  }
#else
  __shared__ __align__(16) __bf16 sAh[64 * LS], sAl[64 * LS];
  __shared__ __align__(16) __bf16 sBh[128 * LS], sBl[128 * LS];

  for (int kk = 0; kk < NK; ++kk) {
    const int k0 = kch + kk * 32;
    __syncthreads();
    for (int it = tid; it < 256; it += 128) {
      int r = it >> 2, c = (it & 3) * 8;
      *(uint4*)&sAh[r * LS + c] = *(const uint4*)&hh[(size_t)r * kH + k0 + c];
      *(uint4*)&sAl[r * LS + c] = *(const uint4*)&hl[(size_t)r * kH + k0 + c];
    }
    for (int it = tid; it < 512; it += 128) {
      int r = it >> 2, c = (it & 3) * 8;
      *(uint4*)&sBh[r * LS + c] = *(const uint4*)&whh[(size_t)(n0 + r) * kH + k0 + c];
      *(uint4*)&sBl[r * LS + c] = *(const uint4*)&whl[(size_t)(n0 + r) * kH + k0 + c];
    }
    __syncthreads();
    compute(sAh, sAl, sBh, sBl);
  }
#endif

  #pragma unroll
  for (int j = 0; j < 8; ++j) {
    #pragma unroll
    for (int r = 0; r < 8; ++r) {
      int b = mb + rA + r;
      int n = n0 + j * 16 + lh;
      __hip_atomic_fetch_add(&pre[(size_t)b * k4H + n], acc[j][r],
                             __ATOMIC_RELAXED, __HIP_MEMORY_SCOPE_AGENT);
    }
  }
}

// ---------------------------------------------------------------- phase 2b
// Gates + state update. Reproduces the reference quirk: i/f/o = sg(sg(p)).
__global__ void k_gates(const float* __restrict__ pre, float* __restrict__ c,
                        __bf16* __restrict__ hh, __bf16* __restrict__ hl,
                        float* __restrict__ out, int t) {
  int idx = blockIdx.x * blockDim.x + threadIdx.x;
  if (idx >= kB * kH) return;
  int b = idx >> 10;
  int j = idx & 1023;
  const float* p = pre + (size_t)b * k4H + j;
  float pi = p[0], pf = p[kH], po = p[2 * kH], pg = p[3 * kH];
  float it = sigf(sigf(pi));
  float ft = sigf(sigf(pf));
  float ot = sigf(sigf(po));
  float gt = tanhf(pg);
  float cn = ft * c[idx] + it * gt;
  float hn = ot * tanhf(cn);
  c[idx] = cn;
  out[(size_t)b * kT * kH + (size_t)t * kH + j] = hn;
  __bf16 hb = (__bf16)hn;
  hh[idx] = hb;
  hl[idx] = (__bf16)(hn - (float)hb);
  if (t == kT - 1) {
    float* hT = out + (size_t)kB * kT * kH;
    float* cT = hT + kB * kH;
    hT[idx] = hn;
    cT[idx] = cn;
  }
}

// ---------------------------------------------------------------- launch
extern "C" void kernel_launch(void* const* d_in, const int* in_sizes, int n_in,
                              void* d_out, int out_size, void* d_ws, size_t ws_size,
                              hipStream_t stream) {
  (void)in_sizes; (void)n_in; (void)out_size; (void)ws_size;
  const float* x  = (const float*)d_in[0];   // [B,T,I]
  const float* Wi = (const float*)d_in[1];   // [4H,I]
  const float* Wh = (const float*)d_in[2];   // [4H,H]
  float* out = (float*)d_out;

  char* ws = (char*)d_ws;
  size_t off = 0;
  auto carve = [&](size_t bytes) -> char* {
    char* p = ws + off;
    off = (off + bytes + 255) & ~(size_t)255;
    return p;
  };
  float*  xW  = (float*) carve((size_t)kM * k4H * sizeof(float));    // 512 MB, [t][b][n]
  __bf16* xh  = (__bf16*)carve((size_t)kM * kI * 2);
  __bf16* xl  = (__bf16*)carve((size_t)kM * kI * 2);
  __bf16* wih = (__bf16*)carve((size_t)k4H * kI * 2);
  __bf16* wil = (__bf16*)carve((size_t)k4H * kI * 2);
  __bf16* whh = (__bf16*)carve((size_t)k4H * kH * 2);
  __bf16* whl = (__bf16*)carve((size_t)k4H * kH * 2);
  __bf16* hh  = (__bf16*)carve((size_t)kB * kH * 2);
  __bf16* hl  = (__bf16*)carve((size_t)kB * kH * 2);
  float*  cst = (float*) carve((size_t)kB * kH * sizeof(float));

  const int nx = kM * kI, nwi = k4H * kI, nwh = k4H * kH, ns = kB * kH;
  k_split<<<(nx  + 255) / 256, 256, 0, stream>>>(x,  xh,  xl,  nx);
  k_split<<<(nwi + 255) / 256, 256, 0, stream>>>(Wi, wih, wil, nwi);
  k_split<<<(nwh + 255) / 256, 256, 0, stream>>>(Wh, whh, whl, nwh);
  k_init_state<<<(ns + 255) / 256, 256, 0, stream>>>(hh, hl, cst, ns);

  k_gemm_xWi<<<dim3(kM / 128, k4H / 128), 128, 0, stream>>>(xh, xl, wih, wil, xW);

  for (int t = 0; t < kT; ++t) {
    float* pre = xW + (size_t)t * kB * k4H;
    k_step_hWh<<<dim3(k4H / 128, 8), 128, 0, stream>>>(hh, hl, whh, whl, pre);
    k_gates<<<(ns + 255) / 256, 256, 0, stream>>>(pre, cst, hh, hl, out, t);
  }
}
